// DigitConvolutionalModel_75179107549292
// MI455X (gfx1250) — compile-verified, matrix operand order checked by reading, not run on hardware
//
#include <hip/hip_runtime.h>
#include <hip/hip_bf16.h>
#include <stdint.h>

// ---------------------------------------------------------------------------
// CDNA5 bf16 WMMA types
// ---------------------------------------------------------------------------
typedef __attribute__((ext_vector_type(16))) __bf16 v16bf;
typedef __attribute__((ext_vector_type(8)))  float  v8f;
typedef int v4i __attribute__((vector_size(16)));   // matches builtin pointee

union FragU {
    uint32_t u[8];
    v16bf    v;
};

// Model dims
#define BATCH   32768
#define K1PAD   704     // 676 padded to 22*32
#define N1      512
#define K2      512
#define N2      512
#define K3      512
#define N3PAD   16
#define NOUT    10

// GEMM tiling
#define BM      128     // block M tile
#define BN      128     // block N tile
#define BK      32      // K step (= WMMA K)
#define LDS_ROW 80      // bytes per LDS tile row: 64B data + 16B pad (bank spread, 16B aligned)

// ---------------------------------------------------------------------------
// Async global->LDS copy of one 16-byte unit (GLOBAL_LOAD_ASYNC_TO_LDS_B128,
// tracked by ASYNCcnt).  Builtin signature (from hipcc diagnostic):
//   (v4i addrspace(1)*, v4i addrspace(3)*, imm offset, imm cpol)
// AS pointers built via integer casts: flat LDS address truncates to the LDS
// offset in the low 32 bits (ISA 10.2 aperture).
// ---------------------------------------------------------------------------
__device__ __forceinline__ void async_copy16(const void* g, void* l)
{
    __builtin_amdgcn_global_load_async_to_lds_b128(
        (__attribute__((address_space(1))) v4i*)(uintptr_t)g,
        (__attribute__((address_space(3))) v4i*)(uint32_t)(uintptr_t)l,
        0, 0);
}

// ---------------------------------------------------------------------------
// Fragment layout per CDNA5 ISA §7.12.2 (16-bit A/B, 16x32):
//   lane l: m/n = l&15, half = l>>4
//   dwords 0..3: K bytes [16*half, 16*half+16)          (contiguous 16B)
//   dwords 4..7: K bytes [32+16*half, 32+16*half+16)    (contiguous 16B)
// ---------------------------------------------------------------------------
__device__ __forceinline__ v16bf lds_frag(const uint8_t* lds, int row, int half)
{
    const uint8_t* p = lds + row * LDS_ROW + 16 * half;
    uint4 lo = *(const uint4*)(p);
    uint4 hi = *(const uint4*)(p + 32);
    FragU f;
    f.u[0] = lo.x; f.u[1] = lo.y; f.u[2] = lo.z; f.u[3] = lo.w;
    f.u[4] = hi.x; f.u[5] = hi.y; f.u[6] = hi.z; f.u[7] = hi.w;
    return f.v;
}

// direct-from-global fragment (row-major [row, K] bf16), used by gemm_out
__device__ __forceinline__ v16bf load_frag_rowmajor(const __bf16* __restrict__ base,
                                                    int row, int ld, int k0, int lane)
{
    const int half = lane >> 4;
    const __bf16* p = base + (size_t)row * ld + k0 + half * 8;
    FragU f;
#pragma unroll
    for (int v = 0; v < 4; ++v) {
        f.u[v]     = *(const uint32_t*)(p + 2 * v);
        f.u[v + 4] = *(const uint32_t*)(p + 16 + 2 * v);
    }
    return f.v;
}

// ---------------------------------------------------------------------------
// 1) conv 3x3 valid + cast to bf16, K-padded to 704 (zeros in 676..703)
// ---------------------------------------------------------------------------
__global__ __launch_bounds__(256) void conv_cast_kernel(
    const float* __restrict__ x, const float* __restrict__ w,
    __bf16* __restrict__ h0)
{
    const int idx = blockIdx.x * blockDim.x + threadIdx.x;
    const int total = BATCH * K1PAD;
    if (idx >= total) return;
    const int b = idx / K1PAD;
    const int p = idx - b * K1PAD;
    float v = 0.0f;
    if (p < 676) {
        const int i = p / 26;
        const int j = p - i * 26;
        const float* xb = x + (size_t)b * 784 + i * 28 + j;
        v = xb[0]  * w[0] + xb[1]  * w[1] + xb[2]  * w[2]
          + xb[28] * w[3] + xb[29] * w[4] + xb[30] * w[5]
          + xb[56] * w[6] + xb[57] * w[7] + xb[58] * w[8];
    }
    h0[idx] = (__bf16)v;
}

// ---------------------------------------------------------------------------
// 2) f32 -> bf16 weight cast with K (and N) zero-padding.
// ---------------------------------------------------------------------------
__global__ __launch_bounds__(256) void cast_pad_bf16_kernel(
    const float* __restrict__ src, __bf16* __restrict__ dst,
    int Nsrc, int Ksrc, int Npad, int Kpad)
{
    const int idx = blockIdx.x * blockDim.x + threadIdx.x;
    const int total = Npad * Kpad;
    if (idx >= total) return;
    const int n = idx / Kpad;
    const int k = idx - n * Kpad;
    float v = (n < Nsrc && k < Ksrc) ? src[(size_t)n * Ksrc + k] : 0.0f;
    dst[idx] = (__bf16)v;
}

// ---------------------------------------------------------------------------
// 3) bf16 WMMA GEMM, LDS double-buffered via async global->LDS DMA.
//    D[M,N] = relu( A[M,K] * Bt[N,K]^T + bias[N] )  (bf16 out)
//    Block: 256 threads = 8 waves, 4(M) x 2(N) wave grid.
//    Block tile: 128M x 128N; wave tile: 32M x 64N -> 8 wmma / K-step.
//    Per K-step: issue async copies for tile k+1, compute tile k,
//    s_wait_asynccnt 0, barrier, swap buffers.
// ---------------------------------------------------------------------------
template <bool RELU>
__global__ __launch_bounds__(256) void gemm_bf16_kernel(
    const __bf16* __restrict__ A,   // [M, K]
    const __bf16* __restrict__ Bt,  // [N, K]
    const float*  __restrict__ bias,// [N]
    __bf16*       __restrict__ D,   // [M, ldd]
    int K, int ldd)
{
    __shared__ __align__(16) uint8_t ldsA[2][BM * LDS_ROW];
    __shared__ __align__(16) uint8_t ldsB[2][BN * LDS_ROW];

    const int tid  = threadIdx.x;
    const int lane = tid & 31;
    const int wave = tid >> 5;
    const int wm = wave & 3;   // 0..3 -> 32-row M slice
    const int wn = wave >> 2;  // 0..1 -> 64-col N slice
    const int l15  = lane & 15;
    const int half = lane >> 4;
    const int m0blk = blockIdx.x * BM;
    const int n0blk = blockIdx.y * BN;

    // cooperative staging: 128 rows x 64B = 512 x 16B units, 2 per thread
    int srow[2], sseg[2];
#pragma unroll
    for (int i = 0; i < 2; ++i) {
        const int u = tid + i * 256;
        srow[i] = u >> 2;       // 4 units per 64B row
        sseg[i] = u & 3;
    }

    auto stage_async = [&](const __bf16* __restrict__ g, int row0, int ld, int k0,
                           uint8_t* lbase) {
#pragma unroll
        for (int i = 0; i < 2; ++i) {
            const void* gsrc = g + (size_t)(row0 + srow[i]) * ld + k0 + sseg[i] * 8;
            async_copy16(gsrc, lbase + srow[i] * LDS_ROW + sseg[i] * 16);
        }
    };

    v8f c[2][4];
#pragma unroll
    for (int i = 0; i < 2; ++i)
#pragma unroll
        for (int t = 0; t < 4; ++t)
            c[i][t] = (v8f){};

    // prologue: stage K-tile 0 into buffer 0
    stage_async(A,  m0blk, K, 0, ldsA[0]);
    stage_async(Bt, n0blk, K, 0, ldsB[0]);
    __builtin_amdgcn_s_wait_asynccnt(0);
    __syncthreads();

    int buf = 0;
    for (int k0 = 0; k0 < K; k0 += BK) {
        const int kn = k0 + BK;
        const bool more = kn < K;
        if (more) {                       // async DMA of next K-tile into buf^1
            stage_async(A,  m0blk, K, kn, ldsA[buf ^ 1]);
            stage_async(Bt, n0blk, K, kn, ldsB[buf ^ 1]);
        }

        // compute current K-tile from LDS (overlaps the DMA above)
        v16bf a0 = lds_frag(ldsA[buf], wm * 32 +      l15, half);
        v16bf a1 = lds_frag(ldsA[buf], wm * 32 + 16 + l15, half);
#pragma unroll
        for (int t = 0; t < 4; ++t) {
            v16bf b = lds_frag(ldsB[buf], wn * 64 + 16 * t + l15, half);
            c[0][t] = __builtin_amdgcn_wmma_f32_16x16x32_bf16(false, a0, false, b, (short)0, c[0][t], false, false);
            c[1][t] = __builtin_amdgcn_wmma_f32_16x16x32_bf16(false, a1, false, b, (short)0, c[1][t], false, false);
        }

        if (more)
            __builtin_amdgcn_s_wait_asynccnt(0);  // own DMAs landed in LDS
        __syncthreads();                          // all waves' DMAs visible
        buf ^= 1;
    }

    // epilogue: C/D layout -> lane owns col n = l&15 (+tile), rows r + 8*half
    float bv[4];
#pragma unroll
    for (int t = 0; t < 4; ++t)
        bv[t] = bias[n0blk + wn * 64 + 16 * t + l15];

#pragma unroll
    for (int i = 0; i < 2; ++i) {
        const int mbase = m0blk + wm * 32 + i * 16 + half * 8;
#pragma unroll
        for (int r = 0; r < 8; ++r) {
            __bf16* drow = D + (size_t)(mbase + r) * ldd + n0blk + wn * 64 + l15;
#pragma unroll
            for (int t = 0; t < 4; ++t) {
                float v = c[i][t][r] + bv[t];
                if (RELU) v = fmaxf(v, 0.0f);
                drow[16 * t] = (__bf16)v;
            }
        }
    }
}

// ---------------------------------------------------------------------------
// 4) output GEMM: [M,512] x [16,512]^T + out_b -> f32 logits [M,10]
// ---------------------------------------------------------------------------
__global__ __launch_bounds__(256) void gemm_out_kernel(
    const __bf16* __restrict__ A,   // [M, 512]
    const __bf16* __restrict__ Bt,  // [16, 512]
    const float*  __restrict__ bias,// [10]
    float*        __restrict__ out) // [M, 10]
{
    const int lane = threadIdx.x & 31;
    const int wave = threadIdx.x >> 5;
    const int m0 = blockIdx.x * 128 + wave * 16;
    const int l15  = lane & 15;
    const int half = lane >> 4;
    const int mrow = m0 + l15;

    v8f c = {};
#pragma unroll 4
    for (int k0 = 0; k0 < K3; k0 += 32) {
        v16bf a = load_frag_rowmajor(A,  mrow, K3, k0, lane);
        v16bf b = load_frag_rowmajor(Bt, l15,  K3, k0, lane);
        c = __builtin_amdgcn_wmma_f32_16x16x32_bf16(false, a, false, b, (short)0, c, false, false);
    }

    if (l15 < NOUT) {
        const float bb = bias[l15];
#pragma unroll
        for (int r = 0; r < 8; ++r) {
            const int m = m0 + half * 8 + r;
            out[(size_t)m * NOUT + l15] = c[r] + bb;
        }
    }
}

// ---------------------------------------------------------------------------
// launch
// ---------------------------------------------------------------------------
extern "C" void kernel_launch(void* const* d_in, const int* in_sizes, int n_in,
                              void* d_out, int out_size, void* d_ws, size_t ws_size,
                              hipStream_t stream)
{
    const float* x      = (const float*)d_in[0];
    const float* conv_w = (const float*)d_in[1];
    const float* fc1_w  = (const float*)d_in[2];
    const float* fc1_b  = (const float*)d_in[3];
    const float* fc2_w  = (const float*)d_in[4];
    const float* fc2_b  = (const float*)d_in[5];
    const float* out_w  = (const float*)d_in[6];
    const float* out_b  = (const float*)d_in[7];
    float* out = (float*)d_out;

    // workspace layout (bf16 buffers), h2 aliases h0 (h0 dead after GEMM1)
    char* ws = (char*)d_ws;
    size_t off = 0;
    __bf16* h0   = (__bf16*)(ws + off); off += (size_t)BATCH * K1PAD * 2;   // 46.1 MB
    __bf16* h1   = (__bf16*)(ws + off); off += (size_t)BATCH * N1 * 2;      // 33.6 MB
    __bf16* w1bf = (__bf16*)(ws + off); off += (size_t)N1 * K1PAD * 2;      // 0.72 MB
    __bf16* w2bf = (__bf16*)(ws + off); off += (size_t)N2 * K2 * 2;         // 0.52 MB
    __bf16* w3bf = (__bf16*)(ws + off); off += (size_t)N3PAD * K3 * 2;      // 16 KB
    __bf16* h2   = h0;                                                      // alias

    // weight casts (tiny)
    {
        int t1 = N1 * K1PAD;
        cast_pad_bf16_kernel<<<(t1 + 255) / 256, 256, 0, stream>>>(fc1_w, w1bf, N1, 676, N1, K1PAD);
        int t2 = N2 * K2;
        cast_pad_bf16_kernel<<<(t2 + 255) / 256, 256, 0, stream>>>(fc2_w, w2bf, N2, K2, N2, K2);
        int t3 = N3PAD * K3;
        cast_pad_bf16_kernel<<<(t3 + 255) / 256, 256, 0, stream>>>(out_w, w3bf, NOUT, K3, N3PAD, K3);
    }

    // conv + cast + pad
    {
        int total = BATCH * K1PAD;
        conv_cast_kernel<<<(total + 255) / 256, 256, 0, stream>>>(x, conv_w, h0);
    }

    // fc1: [B,704] x [512,704]^T -> relu -> h1 bf16
    {
        dim3 grid(BATCH / BM, N1 / BN);
        gemm_bf16_kernel<true><<<grid, 256, 0, stream>>>(h0, w1bf, fc1_b, h1, K1PAD, N1);
    }
    // fc2: [B,512] x [512,512]^T -> relu -> h2 bf16
    {
        dim3 grid(BATCH / BM, N2 / BN);
        gemm_bf16_kernel<true><<<grid, 256, 0, stream>>>(h1, w2bf, fc2_b, h2, K2, N2);
    }
    // out: [B,512] x [16,512]^T -> f32 logits [B,10]
    {
        gemm_out_kernel<<<BATCH / 128, 256, 0, stream>>>(h2, w3bf, out_b, out);
    }
}
